// SparseMoEBlock_9328668967116
// MI455X (gfx1250) — matrix operand order repaired and checked by reading, not compiled, not run on hardware
//
#include <hip/hip_runtime.h>
#include <hip/hip_bf16.h>

// ---------------------------------------------------------------------------
// Sparse MoE block for MI455X (gfx1250, wave32, WMMA).
// Big GEMMs in bf16 WMMA (f32 accum); gating/top-k/softmax in exact fp32.
// A-tiles staged into LDS with global_load_async_to_lds_b128 (ASYNCcnt).
// ---------------------------------------------------------------------------

#define S_TOK 4096   // B*Sq
#define DIM   1024   // D
#define NEXP  8      // E
#define HID   4096   // H
#define NGATE 4      // G
#define KCAP  1024   // tokens per expert = S*CAP/E

typedef __attribute__((ext_vector_type(16))) __bf16          v16bf;
typedef __attribute__((ext_vector_type(8)))  float           v8f;
typedef __attribute__((ext_vector_type(8)))  unsigned short  us8;

union ABfrag { v16bf v; us8 h[2]; };

__device__ inline unsigned short f32_to_bf16(float f) {
  unsigned int u = __float_as_uint(f);
  u += 0x7FFFu + ((u >> 16) & 1u);   // round-to-nearest-even
  return (unsigned short)(u >> 16);
}

__device__ inline float act_gelu(float v) {
  const float c = 0.7978845608028654f;
  return 0.5f * v * (1.0f + tanhf(c * (v + 0.044715f * v * v * v)));
}
__device__ inline float act_silu(float v) { return v / (1.0f + __expf(-v)); }

// ---------------------------------------------------------------------------
// fp32 -> bf16 bulk convert (x -> xbf, reused by all fc1-style GEMMs)
// ---------------------------------------------------------------------------
__global__ void moe_cvt_bf16(const float* __restrict__ in,
                             unsigned short* __restrict__ out, int n4) {
  int t = blockIdx.x * blockDim.x + threadIdx.x;
  if (t < n4) {
    float4 f = ((const float4*)in)[t];
    ushort4 o;
    o.x = f32_to_bf16(f.x); o.y = f32_to_bf16(f.y);
    o.z = f32_to_bf16(f.z); o.w = f32_to_bf16(f.w);
    ((ushort4*)out)[t] = o;
  }
}

// ---------------------------------------------------------------------------
// mean over G gate heads: gw_avg [E,D], gb_avg [E]
// ---------------------------------------------------------------------------
__global__ void moe_prep_gates(const float* __restrict__ gw,
                               const float* __restrict__ gb,
                               float* __restrict__ gw_avg,
                               float* __restrict__ gb_avg) {
  int t = blockIdx.x * blockDim.x + threadIdx.x;
  if (t < NEXP * DIM) {
    float s = 0.f;
    for (int g = 0; g < NGATE; g++) s += gw[g * NEXP * DIM + t];
    gw_avg[t] = s * (1.0f / NGATE);
  }
  if (t < NEXP) {
    float s = 0.f;
    for (int g = 0; g < NGATE; g++) s += gb[g * NEXP + t];
    gb_avg[t] = s * (1.0f / NGATE);
  }
}

// ---------------------------------------------------------------------------
// per-token gate logits (fp32 exact) + softmax -> scoresT [E, S]
// one block per token; wave w handles expert w
// ---------------------------------------------------------------------------
__global__ __launch_bounds__(256) void moe_gate_softmax(
    const float* __restrict__ x, const float* __restrict__ gw,
    const float* __restrict__ gb, float* __restrict__ scoresT) {
  __shared__ float xrow[DIM];
  __shared__ float lg[NEXP];
  const int s = blockIdx.x;
  const int tid = threadIdx.x;
  ((float4*)xrow)[tid] = ((const float4*)(x + (size_t)s * DIM))[tid];
  __syncthreads();
  const int wave = tid >> 5, lane = tid & 31;
  const float* wrow = gw + wave * DIM;
  float p = 0.f;
  for (int d = lane; d < DIM; d += 32) p += xrow[d] * wrow[d];
  for (int o = 16; o > 0; o >>= 1) p += __shfl_down(p, o);
  if (lane == 0) lg[wave] = p + gb[wave];
  __syncthreads();
  if (tid == 0) {
    float m = lg[0];
    for (int e = 1; e < NEXP; e++) m = fmaxf(m, lg[e]);
    float ex[NEXP]; float sum = 0.f;
    for (int e = 0; e < NEXP; e++) { ex[e] = __expf(lg[e] - m); sum += ex[e]; }
    float inv = 1.0f / sum;
    for (int e = 0; e < NEXP; e++) scoresT[(size_t)e * S_TOK + s] = ex[e] * inv;
  }
}

// ---------------------------------------------------------------------------
// capacity head second layer: cap[s,e] = dot(hcap[s], cap_w2[e]) + cap_b2[e]
// ---------------------------------------------------------------------------
__global__ __launch_bounds__(256) void moe_cap_head(
    const float* __restrict__ hcap, const float* __restrict__ w2,
    const float* __restrict__ b2, float* __restrict__ capout) {
  __shared__ float hrow[DIM];
  const int s = blockIdx.x;
  const int tid = threadIdx.x;
  ((float4*)hrow)[tid] = ((const float4*)(hcap + (size_t)s * DIM))[tid];
  __syncthreads();
  const int wave = tid >> 5, lane = tid & 31;
  const float* wrow = w2 + wave * DIM;
  float p = 0.f;
  for (int d = lane; d < DIM; d += 32) p += hrow[d] * wrow[d];
  for (int o = 16; o > 0; o >>= 1) p += __shfl_down(p, o);
  if (lane == 0) capout[(size_t)s * NEXP + wave] = p + b2[wave];
}

// ---------------------------------------------------------------------------
// per-expert exact top-k via full bitonic sort of 4096 (score,idx) pairs.
// one block (1024 threads) per expert. Also emits the one-hot mask output.
// ---------------------------------------------------------------------------
__global__ __launch_bounds__(1024) void moe_topk(
    const float* __restrict__ scoresT, float* __restrict__ gating,
    int* __restrict__ index, float* __restrict__ ones) {
  __shared__ float vals[S_TOK];
  __shared__ int   idxs[S_TOK];
  const int e = blockIdx.x;
  const int tid = threadIdx.x;
  for (int i = tid; i < S_TOK; i += 1024) {
    vals[i] = scoresT[(size_t)e * S_TOK + i];
    idxs[i] = i;
  }
  __syncthreads();
  for (int ksz = 2; ksz <= S_TOK; ksz <<= 1) {
    for (int j = ksz >> 1; j > 0; j >>= 1) {
      for (int i = tid; i < S_TOK; i += 1024) {
        int ixj = i ^ j;
        if (ixj > i) {
          float va = vals[i], vb = vals[ixj];
          int ia = idxs[i], ib = idxs[ixj];
          // "before" in final (descending, tie: lower idx first) order
          bool before = (va > vb) || (va == vb && ia < ib);
          bool dirFwd = ((i & ksz) == 0);
          if (dirFwd ? !before : before) {
            vals[i] = vb; vals[ixj] = va;
            idxs[i] = ib; idxs[ixj] = ia;
          }
        }
      }
      __syncthreads();
    }
  }
  for (int i = tid; i < S_TOK; i += 1024) {
    if (i < KCAP) {
      gating[e * KCAP + i] = vals[i];
      index[e * KCAP + i]  = idxs[i];
    }
    ones[(size_t)idxs[i] * NEXP + e] = (i < KCAP) ? 1.0f : 0.0f;
  }
}

// ---------------------------------------------------------------------------
// bf16 WMMA GEMM:  C[M,N] = act( A[M,K] * W[N,K]^T + bias[N] )
//   A: bf16 row-major (optionally row-gathered); staged to LDS with
//      global_load_async_to_lds_b128 (ASYNCcnt path, no VGPR round-trip)
//   W: fp32 row-major [N,K], fp32->bf16 converted through VGPRs into LDS
// Block tile 256x128, 8 waves of 64x64, K-chunk 64 (2 WMMA K-steps).
// Per chunk per wave: 32 WMMAs vs 32 ds_load_b128 fragment loads.
// ACT: 0 none, 1 gelu-tanh, 2 silu
// OUTMODE: 0 fp32 store, 1 bf16 store, 2 scaled atomic scatter-add (fp32)
// ---------------------------------------------------------------------------
template<int ACT, int OUTMODE>
__global__ __launch_bounds__(256) void moe_gemm_wmma(
    const unsigned short* __restrict__ A, const float* __restrict__ W,
    const float* __restrict__ bias, float* __restrict__ Cf,
    unsigned short* __restrict__ Cbf, int M, int N, int K,
    const int* __restrict__ gather, const int* __restrict__ scatter,
    const float* __restrict__ scale) {
  __shared__ __align__(16) unsigned short Alds[256 * 64];  // 32 KB
  __shared__ __align__(16) unsigned short Wlds[128 * 64];  // 16 KB

  const int tid  = threadIdx.x;
  const int lane = tid & 31;
  const int wave = tid >> 5;
  const int wr   = wave & 3;    // 4 x 64 rows
  const int wc   = wave >> 2;   // 2 x 64 cols
  const int bm   = blockIdx.y * 256;
  const int bn   = blockIdx.x * 128;
  const int half = lane >> 4;
  const int l15  = lane & 15;

  const v8f vzero = {0.f, 0.f, 0.f, 0.f, 0.f, 0.f, 0.f, 0.f};
  v8f acc[4][4];
#pragma unroll
  for (int i = 0; i < 4; i++)
#pragma unroll
    for (int j = 0; j < 4; j++) acc[i][j] = vzero;

  // --- A-tile async staging setup: 256 rows x 64 bf16 = 2048 x 16B chunks.
  // thread covers 8 chunks: iteration it -> row = it*32 + (tid>>3), seg = tid&7
  const int arow0 = tid >> 3;
  const int aseg  = tid & 7;
  unsigned rowByte[8];     // global byte offset of each covered row
  unsigned ldsAddrA[8];    // per-lane LDS byte address (constant over K loop)
  const unsigned ldsA0 = (unsigned)(uintptr_t)&Alds[0];
#pragma unroll
  for (int it = 0; it < 8; ++it) {
    int r = bm + it * 32 + arow0;
    if (gather) r = gather[r];
    rowByte[it]  = (unsigned)((size_t)r * K * 2u);
    ldsAddrA[it] = ldsA0 + (unsigned)((it * 32 + arow0) * 128 + aseg * 16);
  }

  // --- W-tile staging: 128 rows x 64 fp32 -> bf16; 2 threads/row, 32 floats ea
  const int wrow = tid >> 1;
  const int wseg = tid & 1;
  const float* Wrowp = W + (size_t)(bn + wrow) * K + wseg * 32;
  unsigned short* WldsRow = &Wlds[wrow * 64 + wseg * 32];

  for (int kt = 0; kt < K; kt += 64) {
    // 1) issue async global->LDS copies for the A tile (bf16, no conversion)
#pragma unroll
    for (int it = 0; it < 8; ++it) {
      unsigned goff = rowByte[it] + (unsigned)(kt * 2) + (unsigned)(aseg * 16);
      asm volatile("global_load_async_to_lds_b128 %0, %1, %2"
                   :: "v"(ldsAddrA[it]), "v"(goff), "s"(A)
                   : "memory");
    }
    // 2) stage + convert the W tile through VGPRs (overlaps with async copies)
    {
      const float4* src = (const float4*)(Wrowp + kt);
      ushort4 o[8];
#pragma unroll
      for (int q = 0; q < 8; q++) {
        float4 f = src[q];
        o[q].x = f32_to_bf16(f.x); o[q].y = f32_to_bf16(f.y);
        o[q].z = f32_to_bf16(f.z); o[q].w = f32_to_bf16(f.w);
      }
#pragma unroll
      for (int q = 0; q < 4; q++)
        ((uint4*)WldsRow)[q] = *(uint4*)&o[2 * q];
    }
    // 3) own async copies done, then block-wide visibility
    asm volatile("s_wait_asynccnt 0x0" ::: "memory");
    __syncthreads();

    // 4) 2 K-steps x 16 WMMAs
#pragma unroll
    for (int ks = 0; ks < 2; ++ks) {
      ABfrag bfr[4];
#pragma unroll
      for (int j = 0; j < 4; j++) {
        const int n = wc * 64 + j * 16 + l15;
        bfr[j].h[0] = *(const us8*)&Wlds[n * 64 + ks * 32 + half * 8];
        bfr[j].h[1] = *(const us8*)&Wlds[n * 64 + ks * 32 + 16 + half * 8];
      }
#pragma unroll
      for (int i = 0; i < 4; i++) {
        ABfrag afr;
        const int r = wr * 64 + i * 16 + l15;
        afr.h[0] = *(const us8*)&Alds[r * 64 + ks * 32 + half * 8];
        afr.h[1] = *(const us8*)&Alds[r * 64 + ks * 32 + 16 + half * 8];
#pragma unroll
        for (int j = 0; j < 4; j++) {
          acc[i][j] = __builtin_amdgcn_wmma_f32_16x16x32_bf16(
              false, afr.v, false, bfr[j].v, (short)0, acc[i][j], false, false);
        }
      }
    }
    __syncthreads();
  }

  // epilogue
#pragma unroll
  for (int i = 0; i < 4; i++) {
#pragma unroll
    for (int j = 0; j < 4; j++) {
      const int c = bn + wc * 64 + j * 16 + l15;
      const float b = bias[c];
#pragma unroll
      for (int q = 0; q < 8; q++) {
        const int r = bm + wr * 64 + i * 16 + q + half * 8;
        float v = acc[i][j][q] + b;
        if (ACT == 1) v = act_gelu(v);
        else if (ACT == 2) v = act_silu(v);
        if (OUTMODE == 0) {
          Cf[(size_t)r * N + c] = v;
        } else if (OUTMODE == 1) {
          Cbf[(size_t)r * N + c] = f32_to_bf16(v);
        } else {
          const int dst_row = scatter[r];
          atomicAdd(&Cf[(size_t)dst_row * N + c], scale[r] * v);
        }
      }
    }
  }
}

// ---------------------------------------------------------------------------
// host launcher
// ---------------------------------------------------------------------------
extern "C" void kernel_launch(void* const* d_in, const int* in_sizes, int n_in,
                              void* d_out, int out_size, void* d_ws, size_t ws_size,
                              hipStream_t stream) {
  (void)in_sizes; (void)n_in; (void)out_size; (void)ws_size;

  const float* x      = (const float*)d_in[0];
  const float* gate_w = (const float*)d_in[1];
  const float* gate_b = (const float*)d_in[2];
  const float* cap_w1 = (const float*)d_in[3];
  const float* cap_b1 = (const float*)d_in[4];
  const float* cap_w2 = (const float*)d_in[5];
  const float* cap_b2 = (const float*)d_in[6];
  const float* fc1_w  = (const float*)d_in[7];
  const float* fc1_b  = (const float*)d_in[8];
  const float* fc2_w  = (const float*)d_in[9];
  const float* fc2_b  = (const float*)d_in[10];
  const float* sh_w1  = (const float*)d_in[11];
  const float* sh_b1  = (const float*)d_in[12];
  const float* sh_w2  = (const float*)d_in[13];
  const float* sh_b2  = (const float*)d_in[14];

  float* xout   = (float*)d_out;                       // [S, D]
  float* ones   = xout + (size_t)S_TOK * DIM;          // [S, E]
  float* capout = ones + (size_t)S_TOK * NEXP;         // [S, E]

  // workspace carve-out (256B aligned)
  char* w = (char*)d_ws;
  size_t off = 0;
  auto carve = [&](size_t bytes) {
    void* p = w + off;
    off = (off + bytes + 255) & ~(size_t)255;
    return p;
  };
  unsigned short* xbf     = (unsigned short*)carve((size_t)S_TOK * DIM * 2);
  float*          gw_avg  = (float*)carve((size_t)NEXP * DIM * 4);
  float*          gb_avg  = (float*)carve((size_t)NEXP * 4);
  float*          scoresT = (float*)carve((size_t)NEXP * S_TOK * 4);
  float*          gating  = (float*)carve((size_t)NEXP * KCAP * 4);
  int*            index   = (int*)carve((size_t)NEXP * KCAP * 4);
  float*          hcap    = (float*)carve((size_t)S_TOK * DIM * 4);
  unsigned short* hsh     = (unsigned short*)carve((size_t)S_TOK * HID * 2);
  unsigned short* hex     = (unsigned short*)carve((size_t)KCAP * HID * 2);

  const dim3 blk(256);

  // 1) x -> bf16
  moe_cvt_bf16<<<dim3((S_TOK * DIM / 4 + 255) / 256), blk, 0, stream>>>(
      x, xbf, S_TOK * DIM / 4);

  // 2) fold gate heads
  moe_prep_gates<<<dim3((NEXP * DIM + 255) / 256), blk, 0, stream>>>(
      gate_w, gate_b, gw_avg, gb_avg);

  // 3) exact fp32 logits + softmax
  moe_gate_softmax<<<dim3(S_TOK), blk, 0, stream>>>(x, gw_avg, gb_avg, scoresT);

  // 4) per-expert top-k (+ ones mask)
  moe_topk<<<dim3(NEXP), dim3(1024), 0, stream>>>(scoresT, gating, index, ones);

  // 5) capacity head: silu(x @ cap_w1^T + b1) then small dot head
  moe_gemm_wmma<2, 0><<<dim3(DIM / 128, S_TOK / 256), blk, 0, stream>>>(
      xbf, cap_w1, cap_b1, hcap, nullptr, S_TOK, DIM, DIM,
      nullptr, nullptr, nullptr);
  moe_cap_head<<<dim3(S_TOK), blk, 0, stream>>>(hcap, cap_w2, cap_b2, capout);

  // 6) shared expert: gelu(x @ sh_w1^T + b1) @ sh_w2^T + b2 -> xout (base write)
  moe_gemm_wmma<1, 1><<<dim3(HID / 128, S_TOK / 256), blk, 0, stream>>>(
      xbf, sh_w1, sh_b1, nullptr, hsh, S_TOK, HID, DIM,
      nullptr, nullptr, nullptr);
  moe_gemm_wmma<0, 0><<<dim3(DIM / 128, S_TOK / 256), blk, 0, stream>>>(
      hsh, sh_w2, sh_b2, xout, nullptr, S_TOK, DIM, HID,
      nullptr, nullptr, nullptr);

  // 7) experts: gather fc1 -> gelu -> fc2 -> gated atomic scatter into xout
  for (int e = 0; e < NEXP; e++) {
    const float* w1 = fc1_w + (size_t)e * HID * DIM;
    const float* b1 = fc1_b + (size_t)e * HID;
    const float* w2 = fc2_w + (size_t)e * DIM * HID;
    const float* b2 = fc2_b + (size_t)e * DIM;
    const int*   gi = index + (size_t)e * KCAP;
    const float* gs = gating + (size_t)e * KCAP;

    moe_gemm_wmma<1, 1><<<dim3(HID / 128, KCAP / 256), blk, 0, stream>>>(
        xbf, w1, b1, nullptr, hex, KCAP, HID, DIM, gi, nullptr, nullptr);
    moe_gemm_wmma<0, 2><<<dim3(DIM / 128, KCAP / 256), blk, 0, stream>>>(
        hex, w2, b2, xout, nullptr, KCAP, DIM, HID, nullptr, gi, gs);
  }
}
